// LocalRegionEncoder_81870666596703
// MI455X (gfx1250) — compile-verified, matrix operand order checked by reading, not run on hardware
//
#include <hip/hip_runtime.h>
#include <hip/hip_bf16.h>
#include <stdint.h>

// ---------------- problem constants ----------------
#define NPTS   8192      // N points per batch
#define BATCH  4
#define MANCH  2048      // M anchors
#define KNBR   32        // K neighbors
#define DFEAT  16        // feature channels
#define ROWS   (BATCH * MANCH * KNBR)   // 262144 rows of the MLP "matrix"
#define RADIUS2 (0.2f * 0.2f)
#define BN_EPS 1e-5f

typedef __attribute__((ext_vector_type(16))) _Float16 v16h;
typedef __attribute__((ext_vector_type(8)))  float    v8f;

// =====================================================================
// Kernel 1: Furthest Point Sampling — one workgroup per batch.
// Each thread keeps its 8 points and running dmin entirely in registers;
// per-step argmax = wave shuffle reduce + 32-entry LDS reduce.
// Tie-break matches jnp.argmax (first/lowest index wins).
// =====================================================================
__global__ __launch_bounds__(1024) void fps_kernel(const float* __restrict__ coords,
                                                   float* __restrict__ anchors) {
    const int b = blockIdx.x, tid = threadIdx.x;
    const int PT = NPTS / 1024;   // 8 points per thread
    float px[PT], py[PT], pz[PT], dmin[PT];
#pragma unroll
    for (int j = 0; j < PT; ++j) {
        int i = tid + j * 1024;
        const float* c = coords + ((long)b * NPTS + i) * 3;
        px[j] = c[0]; py[j] = c[1]; pz[j] = c[2];
        dmin[j] = 1e10f;
    }
    __shared__ float rb[32];
    __shared__ int   ri[32];
    __shared__ float sp[3];
    __shared__ int   swin;

    // emit index 0 (reference starts at index 0)
    if (tid == 0) {
        sp[0] = px[0]; sp[1] = py[0]; sp[2] = pz[0];
        float* a = anchors + (long)b * MANCH * 3;
        a[0] = px[0]; a[1] = py[0]; a[2] = pz[0];
    }
    __syncthreads();

    for (int s = 1; s < MANCH; ++s) {
        const float ax = sp[0], ay = sp[1], az = sp[2];
        float bd = -1.0f; int bi = NPTS;
#pragma unroll
        for (int j = 0; j < PT; ++j) {
            float dx = px[j] - ax, dy = py[j] - ay, dz = pz[j] - az;
            float d = dx * dx + dy * dy + dz * dz;
            d = fminf(d, dmin[j]);
            dmin[j] = d;
            int i = tid + j * 1024;
            if (d > bd) { bd = d; bi = i; }
        }
        // wave32 reduce (prefer larger d; on tie prefer smaller index)
        for (int off = 16; off; off >>= 1) {
            float od = __shfl_down(bd, off);
            int   oi = __shfl_down(bi, off);
            if (od > bd || (od == bd && oi < bi)) { bd = od; bi = oi; }
        }
        const int wid = tid >> 5;
        if ((tid & 31) == 0) { rb[wid] = bd; ri[wid] = bi; }
        __syncthreads();
        if (wid == 0) {
            bd = rb[tid & 31]; bi = ri[tid & 31];
            for (int off = 16; off; off >>= 1) {
                float od = __shfl_down(bd, off);
                int   oi = __shfl_down(bi, off);
                if (od > bd || (od == bd && oi < bi)) { bd = od; bi = oi; }
            }
            if (tid == 0) swin = bi;
        }
        __syncthreads();
        const int win = swin;
        const int slot = win >> 10, owner = win & 1023;
        if (tid == owner) {
            sp[0] = px[slot]; sp[1] = py[slot]; sp[2] = pz[slot];
            float* a = anchors + ((long)b * MANCH + s) * 3;
            a[0] = px[slot]; a[1] = py[slot]; a[2] = pz[slot];
        }
        __syncthreads();
    }
}

// =====================================================================
// Kernel 2: ball query + gather. One wave per anchor. Writes the padded
// f16 input matrix X0[ROWS][32] directly (channels 0-2 rel-xyz, 3-18
// features, 19-31 zero). Rows are laid out so the GEMM A-fragment load
// is two contiguous b128s per lane.
// =====================================================================
__device__ __forceinline__ void write_row(_Float16* __restrict__ X0, long g, int pos,
                                          float dx, float dy, float dz,
                                          const float* __restrict__ fptr) {
    __align__(16) _Float16 h[32];
    h[0] = (_Float16)dx; h[1] = (_Float16)dy; h[2] = (_Float16)dz;
#pragma unroll
    for (int t = 0; t < DFEAT; ++t) h[3 + t] = (_Float16)fptr[t];
#pragma unroll
    for (int t = 3 + DFEAT; t < 32; ++t) h[t] = (_Float16)0.0f;
    uint4* dst = (uint4*)(X0 + (g * KNBR + pos) * 32);
    const uint4* src = (const uint4*)h;
    dst[0] = src[0]; dst[1] = src[1]; dst[2] = src[2]; dst[3] = src[3];
}

__global__ __launch_bounds__(256) void ball_gather_kernel(const float* __restrict__ coords,
                                                          const float* __restrict__ feats,
                                                          const float* __restrict__ anchors,
                                                          _Float16* __restrict__ X0) {
    const int lane = threadIdx.x & 31;
    const long g = (long)blockIdx.x * 8 + (threadIdx.x >> 5);   // anchor id in [0, B*M)
    const int b = (int)(g >> 11);                                // / MANCH
    const float* a = anchors + g * 3;
    const float ax = a[0], ay = a[1], az = a[2];

    int cnt = 0, firstIdx = -1;
    for (int base = 0; base < NPTS; base += 32) {
        const int i = base + lane;
        const float* c = coords + ((long)b * NPTS + i) * 3;
        const float dx = c[0] - ax, dy = c[1] - ay, dz = c[2] - az;
        const bool in = (dx * dx + dy * dy + dz * dz) < RADIUS2;
        const unsigned mask = (unsigned)__ballot(in);
        if (mask) {
            if (firstIdx < 0) firstIdx = base + __ffs(mask) - 1;
            const int pos = cnt + __popc(mask & ((1u << lane) - 1u));
            if (in && pos < KNBR)
                write_row(X0, g, pos, dx, dy, dz, feats + ((long)b * NPTS + i) * DFEAT);
            cnt += __popc(mask);
            if (cnt >= KNBR) break;
        }
    }
    if (cnt < KNBR) {   // pad with the first hit (PointNet++ semantics)
        const int fi = (firstIdx < 0) ? (NPTS - 1) : firstIdx;
        if (lane >= cnt && lane < KNBR) {
            const float* c = coords + ((long)b * NPTS + fi) * 3;
            write_row(X0, g, lane, c[0] - ax, c[1] - ay, c[2] - az,
                      feats + ((long)b * NPTS + fi) * DFEAT);
        }
    }
}

// =====================================================================
// Kernel 3: pre-swizzle weights into WMMA B-fragment layout (K x N, f16).
// Per (col_tile, k_chunk): 32 lanes x 16 halves, K index per ISA 7.12.2.
// =====================================================================
__device__ __forceinline__ int k_of(int e, int h) {
    return (e < 8) ? (h * 8 + e) : (16 + h * 8 + (e - 8));
}

__global__ void prep_weights(const float* __restrict__ W0, const float* __restrict__ W1,
                             const float* __restrict__ W2, _Float16* __restrict__ bf) {
    for (int idx = threadIdx.x; idx < 14336; idx += blockDim.x) {
        int rel, kchunks, cin; const float* W;
        if (idx < 2048)      { rel = idx;        kchunks = 1; cin = 19; W = W0; }
        else if (idx < 6144) { rel = idx - 2048; kchunks = 2; cin = 64; W = W1; }
        else                 { rel = idx - 6144; kchunks = 2; cin = 64; W = W2; }
        const int e    = rel & 15;
        const int lane = (rel >> 4) & 31;
        const int tk   = rel >> 9;                 // tile*kchunks + kc
        const int kc   = tk % kchunks;
        const int tile = tk / kchunks;
        const int n = tile * 16 + (lane & 15);
        const int k = kc * 32 + k_of(e, lane >> 4);
        bf[idx] = (_Float16)((k < cin) ? W[(long)n * cin + k] : 0.0f);
    }
}

// =====================================================================
// Kernel 4: GEMM via v_wmma_f32_16x16x32_f16. One 16x16 output tile per
// wave, K accumulated over CIN/32 chunks. Writes raw f16 output and a
// deterministic (sum,sumsq) partial per (tile, channel) for the BN pass.
// =====================================================================
template <int CIN, int COUT>
__global__ __launch_bounds__(256) void gemm_stats_kernel(const _Float16* __restrict__ Xin,
                                                         const _Float16* __restrict__ Bfrag,
                                                         _Float16* __restrict__ Yout,
                                                         float2* __restrict__ partials) {
    const int lane = threadIdx.x & 31;
    const int wave = threadIdx.x >> 5;
    constexpr int NT = COUT / 16;
    constexpr int KC = CIN / 32;
    const int tile = blockIdx.x * 8 + wave;
    const int ct = tile % NT;
    const long rt = tile / NT;
    const int r = lane & 15, h = lane >> 4;
    const _Float16* arow = Xin + (rt * 16 + r) * (long)CIN;

    v8f acc = {};
#pragma unroll
    for (int kc = 0; kc < KC; ++kc) {
        union { uint4 u[2]; v16h v; } ua, ub;
        ua.u[0] = *(const uint4*)(arow + kc * 32 + h * 8);         // global_load_b128
        ua.u[1] = *(const uint4*)(arow + kc * 32 + 16 + h * 8);    // global_load_b128
        const uint4* bp = (const uint4*)(Bfrag + (((ct * KC + kc) << 9) + lane * 16));
        ub.u[0] = bp[0]; ub.u[1] = bp[1];
        acc = __builtin_amdgcn_wmma_f32_16x16x32_f16(false, ua.v, false, ub.v,
                                                     (short)0, acc, false, false);
    }

    // D layout: lane holds column n = ct*16 + (lane&15), rows h*8 + j
    const int n = ct * 16 + r;
    const long mbase = rt * 16 + h * 8;
    float ls = 0.0f, lq = 0.0f;
#pragma unroll
    for (int j = 0; j < 8; ++j) {
        const float v = acc[j];
        Yout[(mbase + j) * COUT + n] = (_Float16)v;
        ls += v; lq += v * v;
    }
    // combine the two half-rows (lane & lane+16 share channel n) -> 16-row tile partial
    ls += __shfl_down(ls, 16);
    lq += __shfl_down(lq, 16);
    if (lane < 16) partials[(long)tile * 16 + r] = make_float2(ls, lq);
}

// =====================================================================
// Kernel 5: deterministic per-channel reduction of (sum,sumsq) partials
// + BN scale/shift computation. One block per channel, fixed-order tree.
// =====================================================================
__global__ __launch_bounds__(256) void reduce_stats_kernel(const float2* __restrict__ partials,
                                                           const float* __restrict__ gamma,
                                                           const float* __restrict__ beta,
                                                           float* __restrict__ scsh,
                                                           int ntiles_row, int nt_col,
                                                           float inv_count) {
    const int c = blockIdx.x;
    const int ct = c >> 4, r = c & 15;
    const int cout = nt_col * 16;
    float s = 0.0f, q = 0.0f;
    for (int rt = threadIdx.x; rt < ntiles_row; rt += 256) {
        const float2 p = partials[((long)rt * nt_col + ct) * 16 + r];
        s += p.x; q += p.y;
    }
    __shared__ float as[256], aq[256];
    as[threadIdx.x] = s; aq[threadIdx.x] = q;
    __syncthreads();
    for (int off = 128; off; off >>= 1) {
        if (threadIdx.x < off) {
            as[threadIdx.x] += as[threadIdx.x + off];
            aq[threadIdx.x] += aq[threadIdx.x + off];
        }
        __syncthreads();
    }
    if (threadIdx.x == 0) {
        const float mu  = as[0] * inv_count;
        const float var = aq[0] * inv_count - mu * mu;
        const float sc  = gamma[c] * rsqrtf(var + BN_EPS);
        scsh[c]        = sc;
        scsh[cout + c] = beta[c] - mu * sc;
    }
}

// =====================================================================
// Kernel 6: in-place BN + ReLU over f16 activations (8 halves / thread).
// =====================================================================
__global__ __launch_bounds__(256) void bn_apply_kernel(_Float16* __restrict__ Y,
                                                       const float* __restrict__ scsh,
                                                       int cout, long total8) {
    const long t = (long)blockIdx.x * 256 + threadIdx.x;
    if (t >= total8) return;
    _Float16* p = Y + t * 8;
    const int n0 = (int)((t * 8) & (cout - 1));
    union { uint4 u; _Float16 hh[8]; } d;
    d.u = *(const uint4*)p;
#pragma unroll
    for (int j = 0; j < 8; ++j) {
        const float v = (float)d.hh[j] * scsh[n0 + j] + scsh[cout + n0 + j];
        d.hh[j] = (_Float16)fmaxf(v, 0.0f);
    }
    *(uint4*)p = d.u;
}

// =====================================================================
// Kernel 7: final BN + ReLU + max over K + transposed f32 store.
// One thread per (b, m, c); out[(b*128 + c)*M + m].
// =====================================================================
__global__ __launch_bounds__(256) void bn_max_kernel(const _Float16* __restrict__ Y2,
                                                     const float* __restrict__ scsh,
                                                     float* __restrict__ outAgg) {
    const long t = (long)blockIdx.x * 256 + threadIdx.x;   // [0, B*M*128)
    const int c = (int)(t & 127);
    const long bm = t >> 7;
    const float sc = scsh[c], sh = scsh[128 + c];
    const _Float16* p = Y2 + bm * KNBR * 128 + c;
    float mx = 0.0f;   // ReLU outputs are >= 0 and K >= 1
#pragma unroll 4
    for (int k = 0; k < KNBR; ++k) {
        const float v = fmaxf((float)p[(long)k * 128] * sc + sh, 0.0f);
        mx = fmaxf(mx, v);
    }
    const long b = bm >> 11, m = bm & 2047;
    outAgg[(b * 128 + c) * (long)MANCH + m] = mx;
}

// =====================================================================
// Host-side launch
// =====================================================================
extern "C" void kernel_launch(void* const* d_in, const int* in_sizes, int n_in,
                              void* d_out, int out_size, void* d_ws, size_t ws_size,
                              hipStream_t stream) {
    (void)in_sizes; (void)n_in; (void)out_size; (void)ws_size;
    const float* coords = (const float*)d_in[0];
    const float* feats  = (const float*)d_in[1];
    const float* W0 = (const float*)d_in[2];
    const float* g0 = (const float*)d_in[3];
    const float* b0 = (const float*)d_in[4];
    const float* W1 = (const float*)d_in[5];
    const float* g1 = (const float*)d_in[6];
    const float* b1 = (const float*)d_in[7];
    const float* W2 = (const float*)d_in[8];
    const float* g2 = (const float*)d_in[9];
    const float* b2 = (const float*)d_in[10];

    float* anchors = (float*)d_out;                         // [B, M, 3]
    float* agg     = (float*)d_out + (long)BATCH * MANCH * 3; // [B, 128, M]

    // ---- workspace layout ----
    char* ws = (char*)d_ws;
    _Float16* bfrag = (_Float16*)ws;                        //  28,672 B  (pad 32 KB)
    float*    scsh  = (float*)(ws + 32768);                 //  3 x 256 f (pad 8 KB)
    _Float16* X0 = (_Float16*)(ws + 40960);                 //  ROWS*32 f16 = 16 MB
    _Float16* Y0 = X0 + (long)ROWS * 32;                    //  ROWS*64 f16 = 32 MB
    _Float16* Y1 = Y0 + (long)ROWS * 64;                    //  32 MB
    _Float16* Y2 = Y1 + (long)ROWS * 64;                    //  ROWS*128 f16 = 64 MB
    float2*  part = (float2*)(Y2 + (long)ROWS * 128);       //  up to 131072*16*8 = 16 MB

    const int ntrow = ROWS / 16;                            // 16384 row tiles
    const float invc = 1.0f / (float)ROWS;

    // 1) FPS -> anchors
    fps_kernel<<<BATCH, 1024, 0, stream>>>(coords, anchors);
    // 2) ball query + gather -> X0 (f16, padded Cin=32)
    ball_gather_kernel<<<(BATCH * MANCH) / 8, 256, 0, stream>>>(coords, feats, anchors, X0);
    // 3) weight fragments
    prep_weights<<<1, 256, 0, stream>>>(W0, W1, W2, bfrag);

    // 4) layer 0: 19(->32) -> 64
    gemm_stats_kernel<32, 64><<<(ntrow * 4) / 8, 256, 0, stream>>>(X0, bfrag, Y0, part);
    reduce_stats_kernel<<<64, 256, 0, stream>>>(part, g0, b0, scsh, ntrow, 4, invc);
    bn_apply_kernel<<<(int)(((long)ROWS * 64 / 8 + 255) / 256), 256, 0, stream>>>(
        Y0, scsh, 64, (long)ROWS * 64 / 8);

    // 5) layer 1: 64 -> 64
    gemm_stats_kernel<64, 64><<<(ntrow * 4) / 8, 256, 0, stream>>>(Y0, bfrag + 2048, Y1, part);
    reduce_stats_kernel<<<64, 256, 0, stream>>>(part, g1, b1, scsh + 256, ntrow, 4, invc);
    bn_apply_kernel<<<(int)(((long)ROWS * 64 / 8 + 255) / 256), 256, 0, stream>>>(
        Y1, scsh + 256, 64, (long)ROWS * 64 / 8);

    // 6) layer 2: 64 -> 128, then BN+ReLU+max(K)+transpose
    gemm_stats_kernel<64, 128><<<(ntrow * 8) / 8, 256, 0, stream>>>(Y1, bfrag + 6144, Y2, part);
    reduce_stats_kernel<<<128, 256, 0, stream>>>(part, g2, b2, scsh + 512, ntrow, 8, invc);
    bn_max_kernel<<<(BATCH * MANCH * 128) / 256, 256, 0, stream>>>(Y2, scsh + 512, agg);
}
